// DrugPNNModel_12421045420606
// MI455X (gfx1250) — compile-verified, matrix operand order checked by reading, not run on hardware
//
#include <hip/hip_runtime.h>
#include <hip/hip_bf16.h>

// ---------------------------------------------------------------------------
// DrugPNN forward: gather-embeddings + pairwise-cross features + 3-layer MLP.
// All heavy GEMMs run on v_wmma_f32_16x16x32_bf16 (f32 accumulate).
// ---------------------------------------------------------------------------

typedef __bf16 bf16_t;
typedef __attribute__((ext_vector_type(16))) __bf16 v16bf;
typedef __attribute__((ext_vector_type(8)))  __bf16 v8bf;
typedef __attribute__((ext_vector_type(8)))  float  v8f;

namespace cfg {
constexpr int F   = 32;
constexpr int V   = 1000;
constexpr int D   = 64;
constexpr int B   = 8192;
constexpr int P   = F * (F - 1) / 2;   // 496
constexpr int IN  = F * D + P;         // 2544
constexpr int INP = 2560;              // IN padded to multiple of 32 (K pad)
constexpr int H1  = 1024;
constexpr int H2  = 512;
constexpr int OUT = 1000;
constexpr int OUTP = 1024;             // OUT padded to multiple of 128 (N pad)
}

// ---------------------------------------------------------------------------
// Weight prep: f32 row-major [K][N]  ->  bf16 "column-major" [Npad][Kpad],
// zero-padding both dims so WMMA K-steps and N-tiles need no edge handling.
// ---------------------------------------------------------------------------
__global__ __launch_bounds__(256) void conv_transpose_kernel(
    const float* __restrict__ W, bf16_t* __restrict__ Wt,
    int K, int N, int Kpad, int Npad)
{
    int t = blockIdx.x * 256 + threadIdx.x;
    int total = Kpad * Npad;
    if (t >= total) return;
    int n = t / Kpad;
    int k = t - n * Kpad;
    float v = (k < K && n < N) ? W[(size_t)k * N + n] : 0.0f;
    Wt[(size_t)n * Kpad + k] = (bf16_t)v;
}

// ---------------------------------------------------------------------------
// Feature build: one workgroup per batch sample.
//   phase 1: gather e[32][64] from tables -> LDS (f32) and A (bf16)
//   phase 2: 496 upper-triangle dot products (f32 VALU, ~1 GFLOP total)
//   phase 3: zero the 16 K-pad tail entries
// ---------------------------------------------------------------------------
__global__ __launch_bounds__(256) void build_features_kernel(
    const int* __restrict__ x, const float* __restrict__ emb,
    bf16_t* __restrict__ A)
{
    using namespace cfg;
    __shared__ float e[F][D + 1];   // stride 65: bank-conflict-free column reads
    const int b   = blockIdx.x;
    const int tid = threadIdx.x;
    bf16_t* Arow = A + (size_t)b * INP;

    for (int i = tid; i < F * D; i += 256) {
        int f = i >> 6;
        int d = i & 63;
        int idx = x[b * F + f];
        float v = emb[((size_t)f * V + idx) * D + d];
        e[f][d] = v;
        Arow[i] = (bf16_t)v;
    }
    __syncthreads();

    for (int p = tid; p < P; p += 256) {
        int rem = p, i = 0;
        while (rem >= F - 1 - i) { rem -= F - 1 - i; ++i; }
        int j = i + 1 + rem;
        float s = 0.0f;
#pragma unroll 8
        for (int d = 0; d < D; ++d) s = fmaf(e[i][d], e[j][d], s);
        Arow[F * D + p] = (bf16_t)s;
    }
    if (tid < INP - IN) Arow[IN + tid] = (bf16_t)0.0f;
}

// ---------------------------------------------------------------------------
// bf16 WMMA GEMM:  C[M][ldC] = epilogue(A[M][K] * Bt[Npad][K]^T + bias)
//   block = 256 threads = 8 waves, computes a 128(M) x 256(N) C tile
//   wave (wm,wn) in 2x4 grid computes a 64x64 tile = 4x4 WMMA accumulators
//   -> 8 fragment loads feed 16 WMMAs per K-step (0.5 KB / WMMA of L0 traffic)
//   The compiler software-pipelines the clause-paired b128 loads ~2 K-steps
//   ahead of the WMMAs (observed s_wait_loadcnt staircase), which is the
//   effective prefetch; explicit prefetch instructions only disrupted that.
//   EPI == 0: bias + ReLU, store bf16 (feeds next GEMM)
//   EPI == 1: bias + sigmoid, store f32, only cols < Nlimit
// Fragment layout per CDNA5 ISA (16-bit A, 16x16x32): lane = {half,idx},
// holds row `idx`, K runs {half*8 .. +7} and {16+half*8 .. +7}; B mirrors
// over columns (hence the [N][K] transposed weight layout -> b128 loads).
// ---------------------------------------------------------------------------
union BF16Frag { v16bf v; v8bf h[2]; };

template <int EPI>
__global__ __launch_bounds__(256) void gemm_wmma_kernel(
    const bf16_t* __restrict__ A, const bf16_t* __restrict__ Bt,
    const float* __restrict__ bias, void* __restrict__ Cout,
    int K, int ldC, int Nlimit)
{
    const int tid  = threadIdx.x;
    const int lane = tid & 31;
    const int wave = tid >> 5;
    const int half = lane >> 4;
    const int idx  = lane & 15;
    const int m0 = blockIdx.y * 128 + (wave >> 2) * 64;  // wave row offset
    const int n0 = blockIdx.x * 256 + (wave & 3) * 64;   // wave col offset

    v8f acc[4][4];
#pragma unroll
    for (int mt = 0; mt < 4; ++mt)
#pragma unroll
        for (int nt = 0; nt < 4; ++nt)
#pragma unroll
            for (int r = 0; r < 8; ++r) acc[mt][nt][r] = 0.0f;

    // Per-lane base pointers for the 4 A-rows and 4 B-rows this lane touches.
    const bf16_t* apb[4];
    const bf16_t* bpb[4];
#pragma unroll
    for (int mt = 0; mt < 4; ++mt)
        apb[mt] = A + (size_t)(m0 + mt * 16 + idx) * K + half * 8;
#pragma unroll
    for (int nt = 0; nt < 4; ++nt)
        bpb[nt] = Bt + (size_t)(n0 + nt * 16 + idx) * K + half * 8;

    for (int k0 = 0; k0 < K; k0 += 32) {
        BF16Frag a[4], bfr[4];
#pragma unroll
        for (int mt = 0; mt < 4; ++mt) {
            const bf16_t* ap = apb[mt] + k0;
            a[mt].h[0] = *(const v8bf*)(ap);
            a[mt].h[1] = *(const v8bf*)(ap + 16);
        }
#pragma unroll
        for (int nt = 0; nt < 4; ++nt) {
            const bf16_t* bp = bpb[nt] + k0;
            bfr[nt].h[0] = *(const v8bf*)(bp);
            bfr[nt].h[1] = *(const v8bf*)(bp + 16);
        }
#pragma unroll
        for (int mt = 0; mt < 4; ++mt)
#pragma unroll
            for (int nt = 0; nt < 4; ++nt)
                acc[mt][nt] = __builtin_amdgcn_wmma_f32_16x16x32_bf16(
                    false, a[mt].v, false, bfr[nt].v,
                    (short)0, acc[mt][nt], false, false);
    }

    // Epilogue.  C/D layout: VGPR r -> row 8*half + r, col idx.
#pragma unroll
    for (int mt = 0; mt < 4; ++mt) {
#pragma unroll
        for (int nt = 0; nt < 4; ++nt) {
            const int col = n0 + nt * 16 + idx;
#pragma unroll
            for (int r = 0; r < 8; ++r) {
                const int row = m0 + mt * 16 + half * 8 + r;
                if (EPI == 0) {
                    float v = acc[mt][nt][r] + bias[col];
                    v = fmaxf(v, 0.0f);
                    ((bf16_t*)Cout)[(size_t)row * ldC + col] = (bf16_t)v;
                } else {
                    if (col < Nlimit) {
                        float v = acc[mt][nt][r] + bias[col];
                        float s = 1.0f / (1.0f + __expf(-v));
                        ((float*)Cout)[(size_t)row * ldC + col] = s;
                    }
                }
            }
        }
    }
}

// ---------------------------------------------------------------------------
extern "C" void kernel_launch(void* const* d_in, const int* in_sizes, int n_in,
                              void* d_out, int out_size, void* d_ws, size_t ws_size,
                              hipStream_t stream)
{
    using namespace cfg;
    (void)in_sizes; (void)n_in; (void)out_size; (void)ws_size;

    const int*   x   = (const int*)  d_in[0];
    const float* emb = (const float*)d_in[1];
    const float* W1  = (const float*)d_in[2];
    const float* b1  = (const float*)d_in[3];
    const float* W2  = (const float*)d_in[4];
    const float* b2  = (const float*)d_in[5];
    const float* W3  = (const float*)d_in[6];
    const float* b3  = (const float*)d_in[7];
    float* out = (float*)d_out;

    // Workspace layout (bf16 buffers, 256B aligned): ~74 MB total.
    char* ws = (char*)d_ws;
    size_t off = 0;
    auto carve = [&](size_t bytes) -> void* {
        void* p = ws + off;
        off = (off + bytes + 255) & ~(size_t)255;
        return p;
    };
    bf16_t* Afeat = (bf16_t*)carve((size_t)B * INP  * sizeof(bf16_t));
    bf16_t* W1t   = (bf16_t*)carve((size_t)H1 * INP * sizeof(bf16_t));
    bf16_t* W2t   = (bf16_t*)carve((size_t)H2 * H1  * sizeof(bf16_t));
    bf16_t* W3t   = (bf16_t*)carve((size_t)OUTP * H2 * sizeof(bf16_t));
    bf16_t* Act1  = (bf16_t*)carve((size_t)B * H1  * sizeof(bf16_t));
    bf16_t* Act2  = (bf16_t*)carve((size_t)B * H2  * sizeof(bf16_t));

    // Weight conversion + transpose (every call: launch must be stateless).
    {
        int tot1 = INP * H1, tot2 = H1 * H2, tot3 = H2 * OUTP;
        conv_transpose_kernel<<<(tot1 + 255) / 256, 256, 0, stream>>>(W1, W1t, IN, H1, INP, H1);
        conv_transpose_kernel<<<(tot2 + 255) / 256, 256, 0, stream>>>(W2, W2t, H1, H2, H1,  H2);
        conv_transpose_kernel<<<(tot3 + 255) / 256, 256, 0, stream>>>(W3, W3t, H2, OUT, H2, OUTP);
    }

    // Embedding gather + Gram cross features -> bf16 A [B][INP]
    build_features_kernel<<<B, 256, 0, stream>>>(x, emb, Afeat);

    // MLP: three WMMA GEMMs with fused epilogues.
    gemm_wmma_kernel<0><<<dim3(H1 / 256,  B / 128), 256, 0, stream>>>(
        Afeat, W1t, b1, Act1, INP, H1, H1);
    gemm_wmma_kernel<0><<<dim3(H2 / 256,  B / 128), 256, 0, stream>>>(
        Act1, W2t, b2, Act2, H1, H2, H2);
    gemm_wmma_kernel<1><<<dim3(OUTP / 256, B / 128), 256, 0, stream>>>(
        Act2, W3t, b3, out, H2, OUT, OUT);
}
